// CWICLinear_41729902248305
// MI455X (gfx1250) — compile-verified, matrix operand order checked by reading, not run on hardware
//
#include <hip/hip_runtime.h>
#include <hip/hip_bf16.h>
#include <math.h>

// ---------------- problem constants (from reference) ----------------
#define T_TOK     512          // 2 * 256 tokens
#define IN_F      2048
#define OUT_F     8192
#define NSTRIPES  64
#define SSTRIPE   128
#define TILE_M    128          // token rows per workgroup
#define KCHUNK    32           // bf16 WMMA K per step
#define FLOPS_CONST 16777216.0f  // IN_F * OUT_F

typedef __attribute__((ext_vector_type(16))) __bf16 v16bf;
typedef __attribute__((ext_vector_type(8)))  __bf16 v8bf;
typedef __attribute__((ext_vector_type(4)))  __bf16 v4bf;
typedef __attribute__((ext_vector_type(2)))  __bf16 v2bf;
typedef __attribute__((ext_vector_type(8)))  float  v8f;

union V16U { v16bf v; v8bf h[2]; };

// ---------------- kernel 1: post_mu[o] = sum_i mu[i] * W[i,o] ----------------
__global__ __launch_bounds__(256)
void postmu_kernel(const float* __restrict__ w, const float* __restrict__ mu,
                   float* __restrict__ pm) {
    int o = blockIdx.x * 256 + threadIdx.x;
    float acc = 0.0f;
    #pragma unroll 8
    for (int i = 0; i < IN_F; ++i)
        acc += mu[i] * w[(size_t)i * OUT_F + o];   // coalesced across o
    pm[o] = acc;
}

// ---------------- kernel 2: flops_dense / flops_sparse per token ----------------
__global__ __launch_bounds__(256)
void flops_kernel(const float* __restrict__ x, const float* __restrict__ thr,
                  const float* __restrict__ mu, const float* __restrict__ stdv,
                  float* __restrict__ fd, float* __restrict__ fs) {
    __shared__ float xabs[IN_F];   // |x - mu| for this token (8 KB)
    __shared__ int   red[256];
    const int t   = blockIdx.x;
    const int tid = threadIdx.x;
    for (int i = tid; i < IN_F; i += 256)
        xabs[i] = fabsf(x[(size_t)t * IN_F + i] - mu[i]);
    __syncthreads();
    int cnt = 0;
    for (int n = 0; n < NSTRIPES; ++n) {
        const float* tr = thr + (size_t)n * IN_F;
        for (int i = tid; i < IN_F; i += 256)
            cnt += (xabs[i] > tr[i] * stdv[i]) ? 1 : 0;
    }
    red[tid] = cnt;
    __syncthreads();
    #pragma unroll
    for (int s = 128; s > 0; s >>= 1) {
        if (tid < s) red[tid] += red[tid + s];
        __syncthreads();
    }
    if (tid == 0) {
        fd[t] = FLOPS_CONST;
        fs[t] = FLOPS_CONST * ((float)red[0] / (float)(NSTRIPES * IN_F));
    }
}

// ---------------- kernel 3: striped masked GEMM via bf16 WMMA ----------------
__global__ __launch_bounds__(256)
void cwic_gemm_kernel(const float* __restrict__ x,  const float* __restrict__ w,
                      const float* __restrict__ bias, const float* __restrict__ thr,
                      const float* __restrict__ mu, const float* __restrict__ stdv,
                      const float* __restrict__ post_mu, float* __restrict__ y) {
    // A tile: masked activations, row-major [m][k]  (128 x 32 bf16, 8 KB)
    // B tile: weight stripe, col-major  [col][k]    (128 x 32 bf16, 8 KB)
    __shared__ __align__(16) __bf16 As [TILE_M  * KCHUNK];
    __shared__ __align__(16) __bf16 Bst[SSTRIPE * KCHUNK];

    const int n    = blockIdx.y;                 // stripe 0..63
    const int m0   = blockIdx.x * TILE_M;        // token-row base
    const int tid  = threadIdx.x;
    const int lane = tid & 31;
    const int wave = tid >> 5;                   // 0..7
    const int wm   = (wave & 3) * 32;            // wave row band
    const int wn   = (wave >> 2) * 64;           // wave col band

    v8f acc[2][4] = {};

    const float* thrn = thr + (size_t)n * IN_F;

    for (int k0 = 0; k0 < IN_F; k0 += KCHUNK) {
        // ---- stage A: load x (f32), center, mask, convert bf16 ----
        {
            const int row = tid >> 3;            // 0..31
            const int cg  = tid & 7;             // float4 group (0..7) -> 32 cols
            const int kc  = k0 + cg * 4;
            float4 muv = *(const float4*)(mu   + kc);
            float4 thv = *(const float4*)(thrn + kc);
            float4 sdv = *(const float4*)(stdv + kc);
            const float tx = thv.x * sdv.x, ty = thv.y * sdv.y,
                        tz = thv.z * sdv.z, tw = thv.w * sdv.w;
            #pragma unroll
            for (int r = 0; r < 4; ++r) {
                const int rr = row + r * 32;
                float4 xv = *(const float4*)(x + (size_t)(m0 + rr) * IN_F + kc);
                float c0 = xv.x - muv.x; c0 = (fabsf(c0) > tx) ? c0 : 0.0f;
                float c1 = xv.y - muv.y; c1 = (fabsf(c1) > ty) ? c1 : 0.0f;
                float c2 = xv.z - muv.z; c2 = (fabsf(c2) > tz) ? c2 : 0.0f;
                float c3 = xv.w - muv.w; c3 = (fabsf(c3) > tw) ? c3 : 0.0f;
                v4bf av;
                av[0] = (__bf16)c0; av[1] = (__bf16)c1;
                av[2] = (__bf16)c2; av[3] = (__bf16)c3;
                *(v4bf*)&As[rr * KCHUNK + cg * 4] = av;   // 8B ds_store
            }
        }
        // ---- stage B: load W rows k0..k0+31, stripe cols, cvt bf16, transpose ----
        {
            const int c4 = (lane) * 4;                    // 0..124 (4 cols)
            #pragma unroll
            for (int rp = wave; rp < 16; rp += 8) {       // row pair index
                const int k = rp * 2;
                const float* wp = w + (size_t)(k0 + k) * OUT_F + n * SSTRIPE + c4;
                float4 w0 = *(const float4*)(wp);
                float4 w1 = *(const float4*)(wp + OUT_F);
                v2bf p0; p0[0] = (__bf16)w0.x; p0[1] = (__bf16)w1.x;
                v2bf p1; p1[0] = (__bf16)w0.y; p1[1] = (__bf16)w1.y;
                v2bf p2; p2[0] = (__bf16)w0.z; p2[1] = (__bf16)w1.z;
                v2bf p3; p3[0] = (__bf16)w0.w; p3[1] = (__bf16)w1.w;
                *(v2bf*)&Bst[(c4 + 0) * KCHUNK + k] = p0; // 4B ds_store per column
                *(v2bf*)&Bst[(c4 + 1) * KCHUNK + k] = p1;
                *(v2bf*)&Bst[(c4 + 2) * KCHUNK + k] = p2;
                *(v2bf*)&Bst[(c4 + 3) * KCHUNK + k] = p3;
            }
        }
        // prefetch next K chunk into cache while we compute
        if (k0 + KCHUNK < IN_F) {
            __builtin_prefetch(x + (size_t)(m0 + (tid >> 1)) * IN_F + k0 + KCHUNK, 0, 0);
            __builtin_prefetch(w + (size_t)(k0 + KCHUNK + (tid >> 3)) * OUT_F + n * SSTRIPE, 0, 0);
        }
        __syncthreads();

        // ---- fragment loads (ISA 16-bit A/B lane layout) ----
        // lane L: row/col = L%16 ; K-halves split by L<16 (K 0-7,16-23) vs L>=16 (K 8-15,24-31)
        const int koff = (lane < 16) ? 0 : 8;
        v16bf afr[2], bfr[4];
        #pragma unroll
        for (int i = 0; i < 2; ++i) {
            const int row = wm + i * 16 + (lane & 15);
            V16U u;
            u.h[0] = *(const v8bf*)&As[row * KCHUNK + koff];
            u.h[1] = *(const v8bf*)&As[row * KCHUNK + koff + 16];
            afr[i] = u.v;
        }
        #pragma unroll
        for (int j = 0; j < 4; ++j) {
            const int col = wn + j * 16 + (lane & 15);
            V16U u;
            u.h[0] = *(const v8bf*)&Bst[col * KCHUNK + koff];
            u.h[1] = *(const v8bf*)&Bst[col * KCHUNK + koff + 16];
            bfr[j] = u.v;
        }
        #pragma unroll
        for (int i = 0; i < 2; ++i)
            #pragma unroll
            for (int j = 0; j < 4; ++j)
                acc[i][j] = __builtin_amdgcn_wmma_f32_16x16x32_bf16(
                    /*neg_a=*/false, afr[i], /*neg_b=*/false, bfr[j],
                    /*c_mod=*/(short)0, acc[i][j],
                    /*reuse_a=*/false, /*reuse_b=*/false);
        __syncthreads();
    }

    // ---- epilogue: y = acc + post_mu + bias ----
    // C layout: VGPR r, lanes 0-15 -> M=r, lanes 16-31 -> M=r+8; N = lane%16
    const int rsel = (lane >> 4) * 8;
    #pragma unroll
    for (int i = 0; i < 2; ++i) {
        const int rbase = m0 + wm + i * 16 + rsel;
        #pragma unroll
        for (int j = 0; j < 4; ++j) {
            const int gcol = n * SSTRIPE + wn + j * 16 + (lane & 15);
            const float add = post_mu[gcol] + bias[gcol];
            #pragma unroll
            for (int r = 0; r < 8; ++r)
                y[(size_t)(rbase + r) * OUT_F + gcol] = acc[i][j][r] + add;
        }
    }
}

// ---------------- host launcher ----------------
extern "C" void kernel_launch(void* const* d_in, const int* in_sizes, int n_in,
                              void* d_out, int out_size, void* d_ws, size_t ws_size,
                              hipStream_t stream) {
    const float* x    = (const float*)d_in[0];   // (2,256,2048)
    const float* w    = (const float*)d_in[1];   // (2048,8192)
    const float* bias = (const float*)d_in[2];   // (8192,)
    const float* thr  = (const float*)d_in[3];   // (64,2048)
    const float* mu   = (const float*)d_in[4];   // (2048,)
    const float* stdv = (const float*)d_in[5];   // (2048,)

    float* y  = (float*)d_out;                       // 512*8192
    float* fd = y + (size_t)T_TOK * OUT_F;           // 512
    float* fs = fd + T_TOK;                          // 512
    float* pm = (float*)d_ws;                        // post_mu scratch (8192 f32)

    postmu_kernel<<<OUT_F / 256, 256, 0, stream>>>(w, mu, pm);
    flops_kernel<<<T_TOK, 256, 0, stream>>>(x, thr, mu, stdv, fd, fs);

    dim3 grid(T_TOK / TILE_M, NSTRIPES);             // (4, 64) workgroups
    cwic_gemm_kernel<<<grid, 256, 0, stream>>>(x, w, bias, thr, mu, stdv, pm, y);
}
